// TopKMoE_12017318494674
// MI455X (gfx1250) — matrix operand order repaired and checked by reading, not compiled, not run on hardware
//
#include <hip/hip_runtime.h>
#include <hip/hip_bf16.h>
#include <stdint.h>

typedef __attribute__((ext_vector_type(16))) _Float16 v16h;
typedef __attribute__((ext_vector_type(8)))  float    v8f;

#define TOKENS 8192
#define D_DIM  1024
#define F_DIM  2048
#define E_NUM  8
#define A_NUM  (TOKENS * 2)
#define CAP    2560

// ---------------------------------------------------------------- WMMA helper
__device__ inline v8f wmma_f16f32(v16h a, v16h b, v8f c) {
  return __builtin_amdgcn_wmma_f32_16x16x32_f16(
      /*neg_a=*/false, a, /*neg_b=*/false, b,
      /*c_mod=*/(short)0, c, /*reuse_a=*/false, /*reuse_b=*/false);
}

union HalfVec { v16h v; uint4 u4[2]; };

// A fragment (16x32, f16, row-major, ld = leading dim).
// ISA layout: lane m = lane&15; lanes<16 hold K {0..7,16..23}, lanes>=16 hold {8..15,24..31}.
__device__ inline v16h load_a_frag(const _Float16* __restrict__ base, int row,
                                   int ld, int kk, int lane) {
  int m  = row + (lane & 15);
  int kb = kk + ((lane & 16) ? 8 : 0);
  const _Float16* p = base + (size_t)m * ld + kb;
  HalfVec u;
  u.u4[0] = *(const uint4*)(p);        // k = kb .. kb+7
  u.u4[1] = *(const uint4*)(p + 16);   // k = kb+16 .. kb+23
  return u.v;
}

// B fragment (32x16) for X*W^T: B[k,n] = W[n,k], W row-major (N,K).
// ISA layout: lane n = lane&15; lanes<16 hold K {0..7,16..23}, lanes>=16 hold {8..15,24..31}.
__device__ inline v16h load_b_frag(const _Float16* __restrict__ base, int col,
                                   int ld, int kk, int lane) {
  int n  = col + (lane & 15);
  int kb = kk + ((lane & 16) ? 8 : 0);
  const _Float16* p = base + (size_t)n * ld + kb;
  HalfVec u;
  u.u4[0] = *(const uint4*)(p);
  u.u4[1] = *(const uint4*)(p + 16);
  return u.v;
}

// Same B fragment but converting f32 weights on the fly (fallback when ws is small).
__device__ inline v16h load_b_frag(const float* __restrict__ base, int col,
                                   int ld, int kk, int lane) {
  int n  = col + (lane & 15);
  int kb = kk + ((lane & 16) ? 8 : 0);
  const float* p = base + (size_t)n * ld + kb;
  union { float4 v4[4]; float f[16]; } q;
#pragma unroll
  for (int j = 0; j < 4; ++j) q.v4[j] = ((const float4*)(p + (j >> 1) * 16))[j & 1];
  v16h r;
#pragma unroll
  for (int i = 0; i < 16; ++i) r[i] = (_Float16)q.f[i];
  return r;
}

// ---------------------------------------------------------------- fill / convert
__global__ void fill_u32_kernel(uint32_t* p, uint32_t v, size_t n) {
  size_t i = (size_t)blockIdx.x * blockDim.x + threadIdx.x;
  size_t s = (size_t)gridDim.x * blockDim.x;
  for (; i < n; i += s) p[i] = v;
}

// f32 -> f16, 4 elements per thread iteration (n must be a multiple of 4).
__global__ void cvt_f32_f16_kernel(const float* __restrict__ src,
                                   _Float16* __restrict__ dst, size_t n4) {
  size_t i = (size_t)blockIdx.x * blockDim.x + threadIdx.x;
  size_t s = (size_t)gridDim.x * blockDim.x;
  for (; i < n4; i += s) {
    float4 f = ((const float4*)src)[i];
    union { _Float16 h[4]; uint2 u; } o;
    o.h[0] = (_Float16)f.x; o.h[1] = (_Float16)f.y;
    o.h[2] = (_Float16)f.z; o.h[3] = (_Float16)f.w;
    ((uint2*)dst)[i] = o.u;
  }
}

// ---------------------------------------------------------------- router
// 1 wave per token; 8 waves (256 thr) per block; 1024 blocks.
__global__ __launch_bounds__(256) void router_kernel(
    const float* __restrict__ x, const float* __restrict__ rw,
    const float* __restrict__ rb, int* __restrict__ eidx,
    float* __restrict__ gateArr, float* __restrict__ impPart) {
  int lane = threadIdx.x & 31;
  int wave = threadIdx.x >> 5;
  int t = blockIdx.x * 8 + wave;
  const float* xp = x + (size_t)t * D_DIM;
  float acc[8] = {0.f, 0.f, 0.f, 0.f, 0.f, 0.f, 0.f, 0.f};
  for (int d = lane; d < D_DIM; d += 32) {
    float xv = xp[d];
#pragma unroll
    for (int e = 0; e < 8; ++e) acc[e] += xv * rw[e * D_DIM + d];
  }
#pragma unroll
  for (int off = 16; off > 0; off >>= 1) {
#pragma unroll
    for (int e = 0; e < 8; ++e) acc[e] += __shfl_xor(acc[e], off, 32);
  }
  __shared__ float sp[8][8];
  if (lane == 0) {
    float lg[8];
#pragma unroll
    for (int e = 0; e < 8; ++e) lg[e] = acc[e] + rb[e];
    float mx = lg[0];
#pragma unroll
    for (int e = 1; e < 8; ++e) mx = fmaxf(mx, lg[e]);
    float p[8]; float sum = 0.f;
#pragma unroll
    for (int e = 0; e < 8; ++e) { p[e] = __expf(lg[e] - mx); sum += p[e]; }
    float is = 1.0f / sum;
#pragma unroll
    for (int e = 0; e < 8; ++e) p[e] *= is;
    int e1 = 0; float p1 = p[0];
#pragma unroll
    for (int e = 1; e < 8; ++e) if (p[e] > p1) { p1 = p[e]; e1 = e; }
    int e2 = 0; float p2 = -1.f;
#pragma unroll
    for (int e = 0; e < 8; ++e) if (e != e1 && p[e] > p2) { p2 = p[e]; e2 = e; }
    float den = p1 + p2; den = (den > 1e-9f) ? den : 1e-9f;
    eidx[2 * t]     = e1;       eidx[2 * t + 1]    = e2;
    gateArr[2 * t]  = p1 / den; gateArr[2 * t + 1] = p2 / den;
#pragma unroll
    for (int e = 0; e < 8; ++e) sp[wave][e] = p[e];
  }
  __syncthreads();
  if (threadIdx.x < 8) {
    float s = 0.f;
    for (int w = 0; w < 8; ++w) s += sp[w][threadIdx.x];  // fixed order: deterministic
    impPart[blockIdx.x * 8 + threadIdx.x] = s;
  }
}

__global__ void reduce_imp_kernel(const float* __restrict__ part,
                                  float* __restrict__ importance) {
  int e = threadIdx.x;
  if (e < 8) {
    float s = 0.f;
    for (int b = 0; b < 1024; ++b) s += part[b * 8 + e];  // fixed order
    importance[e] = s / (float)TOKENS;
  }
}

// ---------------------------------------------------------------- ordered scan
// Single block, 1024 threads. 8 expert counters packed 8-bit into one u64.
__global__ __launch_bounds__(1024) void assign_scan_kernel(
    const int* __restrict__ eidx, int* __restrict__ slotArr,
    int* __restrict__ inv, int* __restrict__ counts) {
  __shared__ int wcnt[32][8];
  __shared__ int gbase[8];
  int tid = threadIdx.x, lane = tid & 31, wave = tid >> 5;
  if (tid < 8) gbase[tid] = 0;
  __syncthreads();
  for (int chunk = 0; chunk < A_NUM; chunk += 1024) {
    int a = chunk + tid;
    int e = eidx[a];
    unsigned long long pack = 1ull << (e * 8);
    unsigned long long incl = pack;
#pragma unroll
    for (int d2 = 1; d2 < 32; d2 <<= 1) {
      unsigned long long up = __shfl_up(incl, (unsigned)d2, 32);
      if (lane >= d2) incl += up;
    }
    int myPrefix = (int)((incl - pack) >> (e * 8)) & 0xff;
    if (lane == 31) {
#pragma unroll
      for (int ex = 0; ex < 8; ++ex) wcnt[wave][ex] = (int)(incl >> (ex * 8)) & 0xff;
    }
    __syncthreads();
    int base = gbase[e];
    for (int w = 0; w < wave; ++w) base += wcnt[w][e];
    int pos  = base + myPrefix;
    int slot = (pos < CAP) ? pos : -1;
    slotArr[a] = slot;
    if (slot >= 0) inv[e * CAP + slot] = a;
    __syncthreads();
    if (tid < 8) {
      int tot = 0;
      for (int w = 0; w < 32; ++w) tot += wcnt[w][tid];
      gbase[tid] += tot;
    }
    __syncthreads();
  }
  if (tid < 8) counts[tid] = gbase[tid];
}

// ---------------------------------------------------------------- scatter x -> xb (f16)
__global__ __launch_bounds__(256) void scatter_kernel(
    const float* __restrict__ x, const int* __restrict__ eidx,
    const int* __restrict__ slotArr, _Float16* __restrict__ xb) {
  int a = blockIdx.x;
  int s = slotArr[a];
  if (s < 0) return;
  int e = eidx[a];
  int t = a >> 1;
  int off = blockIdx.y * 256 + threadIdx.x;
  xb[((size_t)e * CAP + s) * D_DIM + off] = (_Float16)x[(size_t)t * D_DIM + off];
}

// ---------------------------------------------------------------- aux loss
__global__ void aux_kernel(const float* __restrict__ importance,
                           const int* __restrict__ counts, float* __restrict__ dst) {
  if (threadIdx.x == 0) {
    float s = 0.f;
    for (int e = 0; e < 8; ++e)
      s += importance[e] * ((float)counts[e] / (float)A_NUM);
    dst[0] = 0.01f * s * (float)E_NUM;
  }
}

// ---------------------------------------------------------------- GEMM1: H = silu(Xb*W1^T) * (Xb*W3^T)
// grid (40,16), block 256 = 8 waves; block tile 64x128, wave tile 32x32.
template <typename BT>
__global__ __launch_bounds__(256) void gemm1_kernel(
    const _Float16* __restrict__ xb, const BT* __restrict__ w1,
    const BT* __restrict__ w3, _Float16* __restrict__ H, int e) {
  const _Float16* Xe  = xb + (size_t)e * CAP * D_DIM;
  const BT*       W1e = w1 + (size_t)e * F_DIM * D_DIM;
  const BT*       W3e = w3 + (size_t)e * F_DIM * D_DIM;
  int lane = threadIdx.x & 31;
  int wave = threadIdx.x >> 5;
  int rowBase = blockIdx.x * 64 + (wave >> 2) * 32;
  int colBase = blockIdx.y * 128 + (wave & 3) * 32;
  v8f c1[2][2] = {};
  v8f c3[2][2] = {};
  for (int kk = 0; kk < D_DIM; kk += 32) {
    v16h a[2], b1[2], b3[2];
#pragma unroll
    for (int mi = 0; mi < 2; ++mi)
      a[mi] = load_a_frag(Xe, rowBase + mi * 16, D_DIM, kk, lane);
#pragma unroll
    for (int ni = 0; ni < 2; ++ni) {
      b1[ni] = load_b_frag(W1e, colBase + ni * 16, D_DIM, kk, lane);
      b3[ni] = load_b_frag(W3e, colBase + ni * 16, D_DIM, kk, lane);
    }
#pragma unroll
    for (int mi = 0; mi < 2; ++mi)
#pragma unroll
      for (int ni = 0; ni < 2; ++ni) {
        c1[mi][ni] = wmma_f16f32(a[mi], b1[ni], c1[mi][ni]);
        c3[mi][ni] = wmma_f16f32(a[mi], b3[ni], c3[mi][ni]);
      }
  }
  // C/D layout: VGPR r -> M = r (lanes 0-15) or 8+r (lanes 16-31); N = lane&15.
#pragma unroll
  for (int mi = 0; mi < 2; ++mi)
#pragma unroll
    for (int ni = 0; ni < 2; ++ni)
#pragma unroll
      for (int r = 0; r < 8; ++r) {
        int m = rowBase + mi * 16 + ((lane & 16) ? (8 + r) : r);
        int n = colBase + ni * 16 + (lane & 15);
        float h1v = c1[mi][ni][r];
        float h3v = c3[mi][ni][r];
        // silu via v_rcp_f32 (1 ulp) instead of IEEE div chain
        float sw  = h1v * __builtin_amdgcn_rcpf(1.0f + __expf(-h1v));
        H[(size_t)m * F_DIM + n] = (_Float16)(sw * h3v);
      }
}

// ---------------------------------------------------------------- GEMM2: out[t] += gate * (H*W2^T)
// grid (40,8), block 256 = 8 waves; fused gated scatter (race-free: unique (t,d) per element,
// experts run sequentially on the stream).
template <typename BT>
__global__ __launch_bounds__(256) void gemm2_kernel(
    const _Float16* __restrict__ H, const BT* __restrict__ w2,
    const int* __restrict__ inv, const float* __restrict__ gateArr,
    float* __restrict__ out, int e) {
  const BT*  W2e  = w2 + (size_t)e * D_DIM * F_DIM;
  const int* invE = inv + e * CAP;
  int lane = threadIdx.x & 31;
  int wave = threadIdx.x >> 5;
  int rowBase = blockIdx.x * 64 + (wave >> 2) * 32;
  int colBase = blockIdx.y * 128 + (wave & 3) * 32;
  v8f c[2][2] = {};
  for (int kk = 0; kk < F_DIM; kk += 32) {
    v16h a[2], b[2];
#pragma unroll
    for (int mi = 0; mi < 2; ++mi)
      a[mi] = load_a_frag(H, rowBase + mi * 16, F_DIM, kk, lane);
#pragma unroll
    for (int ni = 0; ni < 2; ++ni)
      b[ni] = load_b_frag(W2e, colBase + ni * 16, F_DIM, kk, lane);
#pragma unroll
    for (int mi = 0; mi < 2; ++mi)
#pragma unroll
      for (int ni = 0; ni < 2; ++ni)
        c[mi][ni] = wmma_f16f32(a[mi], b[ni], c[mi][ni]);
  }
#pragma unroll
  for (int mi = 0; mi < 2; ++mi)
#pragma unroll
    for (int r = 0; r < 8; ++r) {
      int cs = rowBase + mi * 16 + ((lane & 16) ? (8 + r) : r);
      int a_id = invE[cs];
      if (a_id >= 0) {
        int t = a_id >> 1;
        float g = gateArr[a_id];
#pragma unroll
        for (int ni = 0; ni < 2; ++ni) {
          int d = colBase + ni * 16 + (lane & 15);
          float* o = out + (size_t)t * D_DIM + d;
          *o += g * c[mi][ni][r];
        }
      }
    }
}

// ---------------------------------------------------------------- launch
extern "C" void kernel_launch(void* const* d_in, const int* in_sizes, int n_in,
                              void* d_out, int out_size, void* d_ws, size_t ws_size,
                              hipStream_t stream) {
  const float* x  = (const float*)d_in[0];
  const float* rw = (const float*)d_in[1];
  const float* rb = (const float*)d_in[2];
  const float* w1 = (const float*)d_in[3];
  const float* w3 = (const float*)d_in[4];
  const float* w2 = (const float*)d_in[5];
  float* out = (float*)d_out;

  // workspace layout (all offsets 256B aligned)
  constexpr size_t WEL  = (size_t)E_NUM * F_DIM * D_DIM;   // elements per weight tensor
  constexpr size_t OFF_EIDX = 0;                           // A_NUM int
  constexpr size_t OFF_GATE = 65536;                       // A_NUM float
  constexpr size_t OFF_SLOT = 131072;                      // A_NUM int
  constexpr size_t OFF_CNT  = 196608;                      // 8 int
  constexpr size_t OFF_IMP  = 196864;                      // 8 float
  constexpr size_t OFF_IMPP = 197120;                      // 1024*8 float
  constexpr size_t OFF_INV  = 229888;                      // E*CAP int
  constexpr size_t OFF_XB   = 311808;                      // E*CAP*D f16 (40 MB)
  constexpr size_t OFF_H    = OFF_XB + (size_t)E_NUM * CAP * D_DIM * 2;  // CAP*F f16 (10 MB)
  constexpr size_t WS_NEED  = OFF_H + (size_t)CAP * F_DIM * 2;           // ~52.7 MB
  // optional f16 weight staging (halves L2 weight traffic in the GEMMs)
  constexpr size_t OFF_W1H  = WS_NEED;                     // 32 MB
  constexpr size_t OFF_W3H  = OFF_W1H + WEL * 2;           // 32 MB
  constexpr size_t OFF_W2H  = OFF_W3H + WEL * 2;           // 32 MB
  constexpr size_t WS_BIG   = OFF_W2H + WEL * 2;           // ~149 MB
  if (ws_size < WS_NEED) return;
  const bool useF16W = (ws_size >= WS_BIG);

  uint8_t* ws = (uint8_t*)d_ws;
  int*      eidx    = (int*)(ws + OFF_EIDX);
  float*    gateArr = (float*)(ws + OFF_GATE);
  int*      slotArr = (int*)(ws + OFF_SLOT);
  int*      counts  = (int*)(ws + OFF_CNT);
  float*    imp     = (float*)(ws + OFF_IMP);
  float*    impPart = (float*)(ws + OFF_IMPP);
  int*      inv     = (int*)(ws + OFF_INV);
  _Float16* xb      = (_Float16*)(ws + OFF_XB);
  _Float16* Hbuf    = (_Float16*)(ws + OFF_H);
  _Float16* w1h     = (_Float16*)(ws + OFF_W1H);
  _Float16* w3h     = (_Float16*)(ws + OFF_W3H);
  _Float16* w2h     = (_Float16*)(ws + OFF_W2H);

  // init: zero output accumulator, zero xb (dropped/unfilled slots), inv = -1
  fill_u32_kernel<<<2048, 256, 0, stream>>>((uint32_t*)out, 0u,
                                            (size_t)TOKENS * D_DIM);
  fill_u32_kernel<<<2048, 256, 0, stream>>>((uint32_t*)xb, 0u,
                                            (size_t)E_NUM * CAP * D_DIM / 2);
  fill_u32_kernel<<<64, 256, 0, stream>>>((uint32_t*)inv, 0xFFFFFFFFu,
                                          (size_t)E_NUM * CAP);

  router_kernel<<<1024, 256, 0, stream>>>(x, rw, rb, eidx, gateArr, impPart);
  reduce_imp_kernel<<<1, 32, 0, stream>>>(impPart, imp);
  assign_scan_kernel<<<1, 1024, 0, stream>>>(eidx, slotArr, inv, counts);
  scatter_kernel<<<dim3(A_NUM, D_DIM / 256), 256, 0, stream>>>(x, eidx, slotArr, xb);
  aux_kernel<<<1, 1, 0, stream>>>(imp, counts, out + (size_t)TOKENS * D_DIM);

  if (useF16W) {
    cvt_f32_f16_kernel<<<4096, 256, 0, stream>>>(w1, w1h, WEL / 4);
    cvt_f32_f16_kernel<<<4096, 256, 0, stream>>>(w3, w3h, WEL / 4);
    cvt_f32_f16_kernel<<<4096, 256, 0, stream>>>(w2, w2h, WEL / 4);
    for (int e = 0; e < E_NUM; ++e) {
      gemm1_kernel<_Float16><<<dim3(CAP / 64, F_DIM / 128), 256, 0, stream>>>(
          xb, w1h, w3h, Hbuf, e);
      gemm2_kernel<_Float16><<<dim3(CAP / 64, D_DIM / 128), 256, 0, stream>>>(
          Hbuf, w2h, inv, gateArr, out, e);
    }
  } else {
    for (int e = 0; e < E_NUM; ++e) {
      gemm1_kernel<float><<<dim3(CAP / 64, F_DIM / 128), 256, 0, stream>>>(
          xb, w1, w3, Hbuf, e);
      gemm2_kernel<float><<<dim3(CAP / 64, D_DIM / 128), 256, 0, stream>>>(
          Hbuf, w2, inv, gateArr, out, e);
    }
  }
}
